// RNN1_82789789598360
// MI455X (gfx1250) — matrix operand order repaired and checked by reading, not compiled, not run on hardware
//
#include <hip/hip_runtime.h>
#include <hip/hip_bf16.h>
#include <cstddef>
#include <cstdint>

// ---------------------------------------------------------------------------
// Types for CDNA5 WMMA (wave32): V_WMMA_F32_16X16X32_BF16, TDM descriptors
// ---------------------------------------------------------------------------
typedef __bf16 bf16_t;
typedef __attribute__((ext_vector_type(16))) __bf16 v16bf;
typedef __attribute__((ext_vector_type(8)))  __bf16 v8bf;
typedef __attribute__((ext_vector_type(8)))  float  v8f;
typedef __attribute__((ext_vector_type(4)))  unsigned int v4u;
typedef __attribute__((ext_vector_type(8)))  int v8i;
typedef __attribute__((ext_vector_type(4)))  int v4i;

#define B_  1024
#define S_  200
#define E_  64
#define U_  256
#define G_  1024   // 4*U
#define VP1 100001 // V+1

__device__ __forceinline__ v8f wmma_bf16(v16bf a, v16bf b, v8f c) {
  return __builtin_amdgcn_wmma_f32_16x16x32_bf16(false, a, false, b, (short)0, c,
                                                 false, false);
}

// A-matrix (16x32 bf16, M x K) load from row-major [16 x ldk] buffer.
__device__ __forceinline__ v16bf load_a16(const bf16_t* base, int ldk, int kofs,
                                          int lane) {
  int row = lane & 15;
  int kb  = (lane >> 4) * 16;
  const v8bf* p0 = reinterpret_cast<const v8bf*>(base + row * ldk + kofs + kb);
  const v8bf* p1 = reinterpret_cast<const v8bf*>(base + row * ldk + kofs + kb + 8);
  v8bf lo = *p0, hi = *p1;
  return __builtin_shufflevector(lo, hi, 0, 1, 2, 3, 4, 5, 6, 7,
                                 8, 9, 10, 11, 12, 13, 14, 15);
}

// B-matrix (32x16 bf16, K x N) load from (N,K)-major storage (row n = K contig).
__device__ __forceinline__ v16bf load_b16(const bf16_t* baseNK, int ldk, int n0,
                                          int kofs, int lane) {
  int n  = n0 + (lane & 15);
  int kb = (lane >> 4) * 16;
  return *reinterpret_cast<const v16bf*>(baseNK + (size_t)n * ldk + kofs + kb);
}

__device__ __forceinline__ float sigm(float x) {
  return 1.0f / (1.0f + __expf(-x));
}

// ---------------------------------------------------------------------------
// TDM: stage one recurrence K-chunk (4 gate blocks x 32 rows x 64B) into LDS.
// 3D descriptor, dword units: X=16 dw, Y=32 rows (stride 128 dw),
// Z=4 blocks (stride 256*128 dw). One instruction moves 8KB -> LDS.
// Dynamic descriptor dwords go through readfirstlane (TDM operands are SGPRs).
// ---------------------------------------------------------------------------
__device__ __forceinline__ void tdm_stage_rec(const bf16_t* gtile,
                                              unsigned lds_addr) {
  unsigned long long ga = (unsigned long long)(uintptr_t)gtile;
  unsigned lo = (unsigned)__builtin_amdgcn_readfirstlane((int)(unsigned)ga);
  unsigned hi = (unsigned)__builtin_amdgcn_readfirstlane(
      (int)(((unsigned)(ga >> 32) & 0x01FFFFFFu) | (2u << 30)));  // | type=2
  unsigned la = (unsigned)__builtin_amdgcn_readfirstlane((int)lds_addr);
  v4u g0 = { 1u,    // count=1 (valid, user)
             la,    // LDS byte address
             lo,    // global_addr lo
             hi };  // global_addr hi | type=2
  v8i g1 = { (int)(2u << 16),            // data_size = 4B; no multicast/flags
             (int)(128u << 16),          // tensor_dim0 = 128 dw (bits 79:48 lo)
             (int)(1024u << 16),         // dim0 hi=0 | tensor_dim1 = 1024 rows
             (int)(16u << 16),           // dim1 hi=0 | tile_dim0 = 16 dw
             (int)(32u | (4u << 16)),    // tile_dim1 = 32 | tile_dim2 = 4
             (int)128u,                  // tensor_dim0_stride = 128 dw
             (int)(0x8000u << 16),       // tensor_dim1_stride lo16 = 32768
             0 };                        // tensor_dim1_stride hi = 0
  v4i g2 = { 4, 0, 0, 0 };               // tensor_dim2 = 4 (Z OOB bound)
  v4i g3 = { 0, 0, 0, 0 };
#if __clang_major__ >= 23
  v8i gx = { 0, 0, 0, 0, 0, 0, 0, 0 };
  __builtin_amdgcn_tensor_load_to_lds(g0, g1, g2, g3, gx, 0);
#else
  __builtin_amdgcn_tensor_load_to_lds(g0, g1, g2, g3, 0);
#endif
}

// ---------------------------------------------------------------------------
// f32 -> bf16 conversion kernels (plain + transposing to (N,K)-major)
// ---------------------------------------------------------------------------
__global__ void cvt_kernel(const float* __restrict__ src, bf16_t* __restrict__ dst,
                           int n) {
  for (int i = blockIdx.x * blockDim.x + threadIdx.x; i < n;
       i += gridDim.x * blockDim.x)
    dst[i] = (bf16_t)src[i];
}

__global__ void cvtT_kernel(const float* __restrict__ src, bf16_t* __restrict__ dst,
                            int rows, int cols) {
  int n = rows * cols;
  for (int i = blockIdx.x * blockDim.x + threadIdx.x; i < n;
       i += gridDim.x * blockDim.x) {
    int r = i / cols, c = i - r * cols;
    dst[(size_t)c * rows + r] = (bf16_t)src[i];
  }
}

// ---------------------------------------------------------------------------
// Fused gather + x_proj + LSTM + dense head.
// WG = 16 batch rows, 8 waves; wave w owns hidden slice u in [32w, 32w+32).
// kernT pinned in VGPRs; recT DMA'd per K-chunk by the Tensor Data Mover into
// per-wave double-buffered LDS staging (TENSORcnt pipelined).
// ---------------------------------------------------------------------------
__global__ __launch_bounds__(256) void lstm_kernel(
    const int* __restrict__ inputs, const float* __restrict__ bias,
    const bf16_t* __restrict__ embB,   // (V+1, 64) bf16
    const bf16_t* __restrict__ kernT,  // (1024, 64) bf16 == kernel^T
    const bf16_t* __restrict__ recT,   // (1024, 256) bf16 == rec_kernel^T
    const bf16_t* __restrict__ denseT, // (64, 256) bf16 == dense_w^T
    const float* __restrict__ dense_b,
    float* __restrict__ outp)          // (1024, 64) f32 pre-logits
{
  __shared__ bf16_t xs[16 * 64];                      // step embeddings (A)
  __shared__ bf16_t hs[16 * 256];                     // hidden state (A)
  __shared__ float  sMask[16];
  __shared__ __align__(64) bf16_t bstage[8][2][128 * 32];  // TDM staging

  const int tid  = threadIdx.x;
  const int lane = tid & 31;
  const int w    = tid >> 5;
  const int m0   = blockIdx.x * 16;
  const int u0   = w * 32;
  const int ln15 = lane & 15;
  const int Mofs = (lane >> 4) * 8;

  for (int i = tid; i < 16 * 256; i += 256) hs[i] = (bf16_t)0.0f;

  // LDS byte addresses of this wave's two staging buffers (for TDM),
  // plus the addrspace(3)-provenance base pointer (for ds reads).
  unsigned ldsb[2] = { (unsigned)(uintptr_t)&bstage[w][0][0],
                       (unsigned)(uintptr_t)&bstage[w][1][0] };
  const bf16_t* bbase = &bstage[w][0][0];

  // Tile tl = q*2 + hf covers z columns q*256 + u0 + hf*16 .. +15
  int n0t[8];
#pragma unroll
  for (int q = 0; q < 4; ++q)
#pragma unroll
    for (int hf = 0; hf < 2; ++hf) n0t[q * 2 + hf] = q * 256 + u0 + hf * 16;

  float bv[8];
#pragma unroll
  for (int tl = 0; tl < 8; ++tl) bv[tl] = bias[n0t[tl] + ln15];

  // Pin kernT B-tiles in registers for the whole time loop (8x2x8 = 128 VGPR)
  v16bf Bx[8][2];
#pragma unroll
  for (int tl = 0; tl < 8; ++tl)
#pragma unroll
    for (int kc = 0; kc < 2; ++kc)
      Bx[tl][kc] = load_b16(kernT, 64, n0t[tl], kc * 32, lane);

  float c[2][8], h[2][8];
#pragma unroll
  for (int hf = 0; hf < 2; ++hf)
#pragma unroll
    for (int g = 0; g < 8; ++g) { c[hf][g] = 0.0f; h[hf][g] = 0.0f; }

  const bf16_t* rbase = recT + (size_t)u0 * U_;  // wave's gate-block base

  __syncthreads();

  for (int t = 0; t < S_; ++t) {
    // ---- gather embeddings (16 rows x 64 bf16), 4 elems / thread -------
    {
      int e = tid * 4;
      int r = e >> 6, k = e & 63;
      int idx = inputs[(m0 + r) * S_ + t];
      if (k == 0) sMask[r] = (idx != 0) ? 1.0f : 0.0f;
      *reinterpret_cast<unsigned long long*>(&xs[e]) =
          *reinterpret_cast<const unsigned long long*>(embB + (size_t)idx * E_ + k);
    }
    __syncthreads();  // xs / sMask ready, prior-step hs reads all done

    // ---- z = bias + x@kernel + h@rec for this wave's 8 tiles -----------
    v8f acc[8];
#pragma unroll
    for (int tl = 0; tl < 8; ++tl) {
#pragma unroll
      for (int i = 0; i < 8; ++i) acc[tl][i] = bv[tl];
    }

    // Kick off TDM for recurrence chunk 0 while doing the x-part WMMAs.
    tdm_stage_rec(rbase, ldsb[0]);

#pragma unroll
    for (int kc = 0; kc < 2; ++kc) {
      v16bf A = load_a16(xs, 64, kc * 32, lane);
#pragma unroll
      for (int tl = 0; tl < 8; ++tl) acc[tl] = wmma_bf16(A, Bx[tl][kc], acc[tl]);
    }

    // Recurrence: 8 K-chunks, TDM double-buffered one chunk ahead.
#pragma unroll
    for (int kc = 0; kc < 8; ++kc) {
      if (kc < 7)
        tdm_stage_rec(rbase + (kc + 1) * 32, ldsb[(kc + 1) & 1]);
      if (kc < 7)
        __builtin_amdgcn_s_wait_tensorcnt(1);  // chunk kc's DMA complete
      else
        __builtin_amdgcn_s_wait_tensorcnt(0);
      // Launder an LDS *element offset* (integer): the base pointer keeps its
      // addrspace(3) provenance so reads stay ds_load_b128, while the opaque
      // offset + memory clobber pin them below the tensorcnt wait and stop
      // any folding of TDM-written data.
      unsigned bofs = (kc & 1) ? (unsigned)(128 * 32) : 0u;
      asm volatile("" : "+v"(bofs) : : "memory");
      const bf16_t* bp = bbase + bofs;
      v16bf A = load_a16(hs, 256, kc * 32, lane);
#pragma unroll
      for (int tl = 0; tl < 8; ++tl) {
        // staged layout: local row = q*32 + hf*16 + n, 32 bf16 of K per row
        v16bf Bm = load_b16(bp, 32, (tl >> 1) * 32 + (tl & 1) * 16, 0, lane);
        acc[tl] = wmma_bf16(A, Bm, acc[tl]);
      }
    }
    __syncthreads();  // all waves done reading hs before it is rewritten

    // ---- gates (all-sigmoid LSTM), masked state update, write h --------
#pragma unroll
    for (int hf = 0; hf < 2; ++hf) {
#pragma unroll
      for (int g = 0; g < 8; ++g) {
        const int Mrow = g + Mofs;
        const float mm = sMask[Mrow];
        float iv = sigm(acc[0 * 2 + hf][g]);
        float fv = sigm(acc[1 * 2 + hf][g]);
        float gv = sigm(acc[2 * 2 + hf][g]);
        float ov = sigm(acc[3 * 2 + hf][g]);
        float cn = fv * c[hf][g] + iv * gv;
        float hn = ov * sigm(cn);
        c[hf][g] = mm * cn + (1.0f - mm) * c[hf][g];
        h[hf][g] = mm * hn + (1.0f - mm) * h[hf][g];
        hs[Mrow * 256 + u0 + hf * 16 + ln15] = (bf16_t)h[hf][g];
      }
    }
    __syncthreads();  // hs ready for next step (and xs free to overwrite)
  }

  // ---- out = h @ dense_w + dense_b : (16x256)@(256x64), waves 0..3 -----
  if (w < 4) {
    const int n0 = w * 16;
    float db = dense_b[n0 + ln15];
    v8f a;
#pragma unroll
    for (int i = 0; i < 8; ++i) a[i] = db;
#pragma unroll
    for (int kc = 0; kc < 8; ++kc) {
      v16bf A = load_a16(hs, 256, kc * 32, lane);
      v16bf Bm = load_b16(denseT, 256, n0, kc * 32, lane);
      a = wmma_bf16(A, Bm, a);
    }
#pragma unroll
    for (int g = 0; g < 8; ++g)
      outp[(size_t)(m0 + g + Mofs) * E_ + n0 + ln15] = a[g];
  }
}

// ---------------------------------------------------------------------------
// logits = out @ emb_table.T : (1024 x 64) @ (64 x 100001)
// ---------------------------------------------------------------------------
__global__ __launch_bounds__(256) void logits_kernel(
    const float* __restrict__ outp, const bf16_t* __restrict__ embB,
    float* __restrict__ dout)
{
  __shared__ bf16_t as[16 * 64];

  const int tid  = threadIdx.x;
  const int lane = tid & 31;
  const int w    = tid >> 5;
  const int m0   = blockIdx.y * 16;
  const int ln15 = lane & 15;
  const int Mofs = (lane >> 4) * 8;
  const int kb   = (lane >> 4) * 16;

  {  // stage A tile as bf16 (16 x 64)
    int e = tid * 4;
    int r = e >> 6, k = e & 63;
#pragma unroll
    for (int i = 0; i < 4; ++i)
      as[e + i] = (bf16_t)outp[(size_t)(m0 + r) * E_ + k + i];
  }
  __syncthreads();

  const int base_n = blockIdx.x * 1024 + w * 128;
#pragma unroll
  for (int nt = 0; nt < 8; ++nt) {
    const int n0 = base_n + nt * 16;
    if (n0 >= VP1) break;  // uniform across the wave
    const int n  = n0 + ln15;
    const int nc = (n < VP1) ? n : (VP1 - 1);
    if (nt < 7)
      __builtin_prefetch(embB + (size_t)(nc + 16) * E_, 0, 3);
    v8f a;
#pragma unroll
    for (int i = 0; i < 8; ++i) a[i] = 0.0f;
#pragma unroll
    for (int kc = 0; kc < 2; ++kc) {
      v16bf A = load_a16(as, 64, kc * 32, lane);
      v16bf Bm = *reinterpret_cast<const v16bf*>(embB + (size_t)nc * E_ +
                                                 kc * 32 + kb);
      a = wmma_bf16(A, Bm, a);
    }
    if (n < VP1) {
#pragma unroll
      for (int g = 0; g < 8; ++g)
        dout[(size_t)(m0 + g + Mofs) * VP1 + n] = a[g];
    }
  }
}

// ---------------------------------------------------------------------------
// Host launcher
// ---------------------------------------------------------------------------
extern "C" void kernel_launch(void* const* d_in, const int* in_sizes, int n_in,
                              void* d_out, int out_size, void* d_ws, size_t ws_size,
                              hipStream_t stream) {
  const int*   inputs  = (const int*)d_in[0];
  const float* emb     = (const float*)d_in[1];
  const float* kern    = (const float*)d_in[2];
  const float* rec     = (const float*)d_in[3];
  const float* bias    = (const float*)d_in[4];
  const float* dense_w = (const float*)d_in[5];
  const float* dense_b = (const float*)d_in[6];
  float*       dout    = (float*)d_out;

  char* ws = (char*)d_ws;
  auto al = [](size_t x) { return (x + 255) & ~(size_t)255; };
  size_t off = 0;
  bf16_t* embB   = (bf16_t*)(ws + off); off += al((size_t)VP1 * E_ * sizeof(bf16_t));
  bf16_t* kernT  = (bf16_t*)(ws + off); off += al((size_t)G_ * E_ * sizeof(bf16_t));
  bf16_t* recT   = (bf16_t*)(ws + off); off += al((size_t)G_ * U_ * sizeof(bf16_t));
  bf16_t* denseT = (bf16_t*)(ws + off); off += al((size_t)E_ * U_ * sizeof(bf16_t));
  float*  outp   = (float*)(ws + off);  off += al((size_t)B_ * E_ * sizeof(float));

  cvt_kernel<<<1024, 256, 0, stream>>>(emb, embB, VP1 * E_);
  cvtT_kernel<<<256, 256, 0, stream>>>(kern, kernT, E_, G_);
  cvtT_kernel<<<256, 256, 0, stream>>>(rec, recT, U_, G_);
  cvtT_kernel<<<64, 256, 0, stream>>>(dense_w, denseT, U_, E_);

  lstm_kernel<<<B_ / 16, 256, 0, stream>>>(inputs, bias, embB, kernT, recT,
                                           denseT, dense_b, outp);

  dim3 lg((VP1 + 1023) / 1024, B_ / 16);
  logits_kernel<<<lg, 256, 0, stream>>>(outp, embB, dout);
}
